// LorentzSelfAttention_26456998543899
// MI455X (gfx1250) — matrix operand order repaired and checked by reading, not compiled, not run on hardware
//
#include <hip/hip_runtime.h>

// ---------------------------------------------------------------------------
// LorentzSelfAttention for MI455X (gfx1250, wave32, WMMA bf16 16x16x32).
//
// B=4, S=2048, D=1024.  FLOPs ~ 69G (QK^T + PV), bytes ~ 64MB -> compute
// bound; run matmuls on v_wmma_f32_16x16x32_bf16.  Softmax shift AND the
// softmax denominator cancel against the final L2 row-normalization, so the
// attention kernel is a pure exp-of-scores streaming GEMM-GEMM.
// exp() uses the raw v_exp_f32 (__builtin_amdgcn_exp2f): |arg| << 126 always.
// ---------------------------------------------------------------------------

#define BATCH 4
#define SEQ   2048
#define DIM   1024
#define NPAIR (DIM / 2)

typedef __attribute__((ext_vector_type(16))) __bf16 v16bf;
typedef __attribute__((ext_vector_type(8)))  __bf16 v8bf;
typedef __attribute__((ext_vector_type(8)))  float  v8f;

// --------------------------- rotation kernels ------------------------------
// Kr[b][s][d]  = Givens(rot_w[1]) applied to Q row s   (row-major, bf16)
__global__ __launch_bounds__(128) void rotate_k_kernel(
    const float* __restrict__ Q, const float* __restrict__ rot,
    __bf16* __restrict__ Kr)
{
    int p = blockIdx.x * 128 + threadIdx.x;   // pair index 0..511
    int s = blockIdx.y;
    int b = blockIdx.z;
    float g0 = rot[1 * DIM + 2 * p], g1 = rot[1 * DIM + 2 * p + 1];
    float inv = rsqrtf(g0 * g0 + g1 * g1);
    float c = g0 * inv, sn = g1 * inv;
    const float* q = Q + ((size_t)b * SEQ + s) * DIM + 2 * p;
    float x0 = q[0], x1 = q[1];
    __bf16* o = Kr + ((size_t)b * SEQ + s) * DIM + 2 * p;
    o[0] = (__bf16)(c * x0 - sn * x1);
    o[1] = (__bf16)(c * x1 + sn * x0);
}

// VrT[b][d][s] = Givens(rot_w[2]) applied to Q, stored TRANSPOSED (bf16) so
// the PV B-fragment (16 contiguous K values per lane) is a contiguous load.
__global__ __launch_bounds__(128) void rotate_v_kernel(
    const float* __restrict__ Q, const float* __restrict__ rot,
    __bf16* __restrict__ VrT)
{
    int s = blockIdx.x * 128 + threadIdx.x;
    int p = blockIdx.y;
    int b = blockIdx.z;
    float g0 = rot[2 * DIM + 2 * p], g1 = rot[2 * DIM + 2 * p + 1];
    float inv = rsqrtf(g0 * g0 + g1 * g1);
    float c = g0 * inv, sn = g1 * inv;
    const float* q = Q + ((size_t)b * SEQ + s) * DIM + 2 * p;
    float x0 = q[0], x1 = q[1];
    VrT[((size_t)b * DIM + 2 * p) * SEQ + s]     = (__bf16)(c * x0 - sn * x1);
    VrT[((size_t)b * DIM + 2 * p + 1) * SEQ + s] = (__bf16)(c * x1 + sn * x0);
}

// --------------------------- fused attention -------------------------------
#define LDA 1032   // padded ldsA row stride (bf16): 2064B = 516 dwords, %64=4
#define LDP 72     // padded pTile row stride (bf16): 144B  =  36 dwords, %64=36

__global__ __launch_bounds__(128) void attn_kernel(
    const float* __restrict__ Q, const float* __restrict__ rot,
    const float* __restrict__ scale_p,
    const __bf16* __restrict__ Kr, const __bf16* __restrict__ VrT,
    float* __restrict__ Out)
{
    __shared__ float  cs0[NPAIR * 2];     // 4KB   q-rotation coefficients
    __shared__ __bf16 ldsA[16 * LDA];     // 33KB  rotated queries (bf16)
    __shared__ __bf16 pTile[16 * LDP];    // 2.3KB exp(score) tile, 16q x 64k
    __shared__ float  rowsq[16];

    const int tid  = threadIdx.x;
    const int wave = tid >> 5;
    const int lane = tid & 31;
    const int ln16 = lane & 15;
    const int h    = lane >> 4;

    const int b  = blockIdx.y;
    const int q0 = blockIdx.x * 16;

    const float a2 = (2.0f / scale_p[0]) * 1.44269504088896340736f; // /ln2

    // ---- stage q-rotation coefficients (rot_w row 0) ----
    for (int i = tid; i < NPAIR; i += 128) {
        float g0 = rot[2 * i], g1 = rot[2 * i + 1];
        float inv = rsqrtf(g0 * g0 + g1 * g1);
        cs0[2 * i] = g0 * inv;
        cs0[2 * i + 1] = g1 * inv;
    }
    if (tid < 16) rowsq[tid] = 0.0f;
    __syncthreads();

    // ---- stage rotated queries (16 x 1024, bf16) into LDS ----
    for (int i = tid; i < 16 * NPAIR; i += 128) {
        int r = i >> 9;            // row 0..15
        int p = i & (NPAIR - 1);   // pair 0..511
        float c = cs0[2 * p], sn = cs0[2 * p + 1];
        const float* qp = Q + ((size_t)b * SEQ + q0 + r) * DIM + 2 * p;
        float x0 = qp[0], x1 = qp[1];
        ldsA[r * LDA + 2 * p]     = (__bf16)(c * x0 - sn * x1);
        ldsA[r * LDA + 2 * p + 1] = (__bf16)(c * x1 + sn * x0);
    }
    __syncthreads();

    // ---- streaming attention: O[16 x 1024], wave owns 256 columns ----
    v8f acc[16];
#pragma unroll
    for (int i = 0; i < 16; ++i) acc[i] = (v8f){};

    const __bf16* KrB = Kr + (size_t)b * SEQ * DIM;
    const __bf16* VtB = VrT + (size_t)b * DIM * SEQ;

    for (int kb = 0; kb < SEQ / 64; ++kb) {
        // Phase 1: this wave computes scores for 16 keys of the 64-key block.
        const int key = kb * 64 + wave * 16 + ln16;     // B-frag column
        const __bf16* krow = KrB + (size_t)key * DIM;
        __builtin_prefetch(krow + 64 * DIM, 0, 3);      // next key block
        v8f sc = (v8f){};
#pragma unroll 4
        for (int dblk = 0; dblk < DIM / 32; ++dblk) {
            // A fragment: row ln16, 16 contiguous K starting at h*16
            const __bf16* ab = &ldsA[ln16 * LDA + dblk * 32 + h * 16];
            v8bf lo = *(const v8bf*)ab;
            v8bf hi = *(const v8bf*)(ab + 8);
            v16bf afrag;
#pragma unroll
            for (int j = 0; j < 8; ++j) { afrag[j] = lo[j]; afrag[j + 8] = hi[j]; }
            // B fragment: key row, 16 contiguous d starting at h*16
            v16bf bfrag = *(const v16bf*)(krow + dblk * 32 + h * 16);
            sc = __builtin_amdgcn_wmma_f32_16x16x32_bf16(
                false, afrag, false, bfrag, (short)0, sc, false, false);
        }
        // p = exp2(score * 2*log2e/scale); raw v_exp_f32 — |arg| <~ 25 always,
        // so no need for libm's overflow/denorm guard sequence.
#pragma unroll
        for (int r = 0; r < 8; ++r) {
            float p = __builtin_amdgcn_exp2f(sc[r] * a2);
            pTile[(r + 8 * h) * LDP + wave * 16 + ln16] = (__bf16)p;
        }
        __syncthreads();

        // Phase 2: O += P_tile @ V for this wave's 256 output columns.
#pragma unroll
        for (int kk = 0; kk < 2; ++kk) {
            const __bf16* pb = &pTile[ln16 * LDP + kk * 32 + h * 16];
            v8bf lo = *(const v8bf*)pb;
            v8bf hi = *(const v8bf*)(pb + 8);
            v16bf pf;
#pragma unroll
            for (int j = 0; j < 8; ++j) { pf[j] = lo[j]; pf[j + 8] = hi[j]; }
            const int kbase = kb * 64 + kk * 32 + h * 16;
#pragma unroll
            for (int nt = 0; nt < 16; ++nt) {
                int n = wave * 256 + nt * 16 + ln16;
                v16bf vf = *(const v16bf*)(VtB + (size_t)n * SEQ + kbase);
                acc[nt] = __builtin_amdgcn_wmma_f32_16x16x32_bf16(
                    false, pf, false, vf, (short)0, acc[nt], false, false);
            }
        }
        __syncthreads();
    }

    // ---- epilogue: per-row L2 normalization ----
    float part[8];
#pragma unroll
    for (int r = 0; r < 8; ++r) part[r] = 0.0f;
#pragma unroll
    for (int nt = 0; nt < 16; ++nt)
#pragma unroll
        for (int r = 0; r < 8; ++r) part[r] += acc[nt][r] * acc[nt][r];
#pragma unroll
    for (int r = 0; r < 8; ++r) atomicAdd(&rowsq[r + 8 * h], part[r]);
    __syncthreads();

    float* outB = Out + ((size_t)b * SEQ + q0) * DIM;
#pragma unroll
    for (int r = 0; r < 8; ++r) {
        int m = r + 8 * h;
        float inv = 1.0f / sqrtf(fmaxf(rowsq[m], 1e-8f));
#pragma unroll
        for (int nt = 0; nt < 16; ++nt)
            outB[(size_t)m * DIM + wave * 256 + nt * 16 + ln16] = acc[nt][r] * inv;
    }
}

// ------------------------------- launch ------------------------------------
extern "C" void kernel_launch(void* const* d_in, const int* in_sizes, int n_in,
                              void* d_out, int out_size, void* d_ws, size_t ws_size,
                              hipStream_t stream) {
    const float* Q     = (const float*)d_in[0];
    const float* rot   = (const float*)d_in[1];
    const float* scale = (const float*)d_in[2];
    // d_in[3] (bias) provably cancels in softmax + L2 normalization.

    __bf16* Kr  = (__bf16*)d_ws;                                         // 16 MB
    __bf16* VrT = (__bf16*)((char*)d_ws + (size_t)BATCH * SEQ * DIM * 2); // 16 MB
    float*  Out = (float*)d_out;

    rotate_k_kernel<<<dim3(NPAIR / 128, SEQ, BATCH), 128, 0, stream>>>(Q, rot, Kr);
    rotate_v_kernel<<<dim3(SEQ / 128, NPAIR, BATCH), 128, 0, stream>>>(Q, rot, VrT);
    attn_kernel<<<dim3(SEQ / 16, BATCH), 128, 0, stream>>>(Q, rot, scale, Kr, VrT, Out);
}
